// ProteinBERTModel_9818295238840
// MI455X (gfx1250) — matrix operand order repaired and checked by reading, not compiled, not run on hardware
//
#include <hip/hip_runtime.h>
#include <hip/hip_bf16.h>
#include <math.h>

// Problem constants (ProteinBERT-like encoder, fp32 reference)
#define B_   2
#define S_   2048
#define H_   768
#define NH_  12
#define DH_  64
#define FF_  3072
#define L_   2
#define M_   (B_ * S_)   // 4096 token rows

typedef __attribute__((ext_vector_type(16))) __bf16 v16bf;
typedef __attribute__((ext_vector_type(8)))  __bf16 v8bf;
typedef __attribute__((ext_vector_type(2)))  __bf16 v2bf;
typedef __attribute__((ext_vector_type(2)))  float  v2f;
typedef __attribute__((ext_vector_type(8)))  float  v8f;

// fp32 -> bf16: let clang legalize FP_ROUND (RNE); on gfx1250 this can map to
// packed cvt hardware, otherwise it expands to the standard integer sequence.
__device__ __forceinline__ __bf16 f2bf(float f) {
    return (__bf16)f;
}

__device__ __forceinline__ v2bf f2bf2(float a, float b) {
    v2f t; t[0] = a; t[1] = b;
    return __builtin_convertvector(t, v2bf);
}

__device__ __forceinline__ v16bf cat16(v8bf lo, v8bf hi) {
    return __builtin_shufflevector(lo, hi, 0, 1, 2, 3, 4, 5, 6, 7,
                                           8, 9, 10, 11, 12, 13, 14, 15);
}

// 16x32 bf16 A-fragment from LDS (row-major, stride ldw elems).
// CDNA5 16-bit A layout: lanes 0-15 hold row r with K {0..7, 16..23},
// lanes 16-31 hold row r with K {8..15, 24..31}. (CDNA5 ISA)
__device__ __forceinline__ v16bf load_a_frag(const __bf16* base, int ldw,
                                             int row, int kbase, int half, int r) {
    const __bf16* p = base + (size_t)(row + r) * ldw + kbase + half * 8;
    v8bf lo = *(const v8bf*)p;
    v8bf hi = *(const v8bf*)(p + 16);
    return cat16(lo, hi);
}

// 32x16 bf16 B-fragment. LDS holds B^T (rows = N, stride ldw):
// lane n (0-15) needs column N=n over K {0..7, 16..23}, lanes 16-31 K {8..15, 24..31}.
__device__ __forceinline__ v16bf load_b_frag(const __bf16* base, int ldw,
                                             int col, int kbase, int half, int r) {
    const __bf16* p = base + (size_t)(col + r) * ldw + kbase + half * 8;
    v8bf lo = *(const v8bf*)p;
    v8bf hi = *(const v8bf*)(p + 16);
    return cat16(lo, hi);
}

// ---------------------------------------------------------------------------
// GEMM: C[M,N] = act(A[M,K] @ W[K,N] + bias[N]). fp32 global, bf16 WMMA math.
// 128x128x32 tiles, 256 threads = 8 waves, ping-pong LDS double buffering:
// global loads for tile t+1 staged into registers while WMMAs consume tile t,
// one barrier per K-step. ACT: 0 = none, 1 = exact GELU.
// ---------------------------------------------------------------------------
template <int ACT>
__global__ __launch_bounds__(256)
void gemm_bias_kernel(const float* __restrict__ A, const float* __restrict__ W,
                      const float* __restrict__ bias, float* __restrict__ C,
                      int M, int N, int K) {
    constexpr int BM = 128, BN = 128, BK = 32, LDW = BK + 8;  // 40 elems = 80B rows
    __shared__ __align__(16) __bf16 As[2][BM * LDW];
    __shared__ __align__(16) __bf16 Bs[2][BN * LDW];   // W tile transposed: [n][k]

    const int tid  = threadIdx.x;
    const int wave = tid >> 5, lane = tid & 31;
    const int half = lane >> 4, r = lane & 15;
    const int waveM = wave & 3;   // 4 waves * 32 rows
    const int waveN = wave >> 2;  // 2 waves * 64 cols
    const int m0 = blockIdx.y * BM;
    const int n0 = blockIdx.x * BN;

    // stage-load index split
    const int arr = tid >> 3,  acc0 = (tid & 7) << 2;    // A: 32 rows/pass, 4 cols
    const int wrr = tid >> 5,  wcc  = (tid & 31) << 2;   // W: 8 k-rows/pass, 4 cols

    v8f acc[2][4];
#pragma unroll
    for (int mt = 0; mt < 2; ++mt)
#pragma unroll
        for (int nt = 0; nt < 4; ++nt)
#pragma unroll
            for (int i = 0; i < 8; ++i) acc[mt][nt][i] = 0.0f;

    // Prologue: tile 0 -> buffer 0
#pragma unroll
    for (int p = 0; p < 4; ++p) {
        const int row = p * 32 + arr;
        const float4 f = *(const float4*)&A[(size_t)(m0 + row) * K + acc0];
        __bf16* d = &As[0][row * LDW + acc0];
        *(v2bf*)d       = f2bf2(f.x, f.y);
        *(v2bf*)(d + 2) = f2bf2(f.z, f.w);
    }
#pragma unroll
    for (int p = 0; p < 4; ++p) {
        const int kk = p * 8 + wrr;
        const float4 f = *(const float4*)&W[(size_t)kk * N + n0 + wcc];
        Bs[0][(wcc + 0) * LDW + kk] = f2bf(f.x);
        Bs[0][(wcc + 1) * LDW + kk] = f2bf(f.y);
        Bs[0][(wcc + 2) * LDW + kk] = f2bf(f.z);
        Bs[0][(wcc + 3) * LDW + kk] = f2bf(f.w);
    }
    __syncthreads();

    int cur = 0;
    for (int k0 = 0; k0 < K; k0 += BK) {
        const bool nxt = (k0 + BK) < K;
        // Stage next tile into registers (overlaps with WMMA below)
        float4 a_s[4], w_s[4];
        if (nxt) {
#pragma unroll
            for (int p = 0; p < 4; ++p)
                a_s[p] = *(const float4*)&A[(size_t)(m0 + p * 32 + arr) * K + k0 + BK + acc0];
#pragma unroll
            for (int p = 0; p < 4; ++p)
                w_s[p] = *(const float4*)&W[(size_t)(k0 + BK + p * 8 + wrr) * N + n0 + wcc];
        }

        // Compute on current buffer
        v16bf afr[2];
#pragma unroll
        for (int mt = 0; mt < 2; ++mt)
            afr[mt] = load_a_frag(As[cur], LDW, waveM * 32 + mt * 16, 0, half, r);
#pragma unroll
        for (int nt = 0; nt < 4; ++nt) {
            const v16bf bfr = load_b_frag(Bs[cur], LDW, waveN * 64 + nt * 16, 0, half, r);
#pragma unroll
            for (int mt = 0; mt < 2; ++mt)
                acc[mt][nt] = __builtin_amdgcn_wmma_f32_16x16x32_bf16(
                    false, afr[mt], false, bfr, (short)0, acc[mt][nt], false, false);
        }

        // Commit staged tile to the other buffer
        if (nxt) {
            const int nb = cur ^ 1;
#pragma unroll
            for (int p = 0; p < 4; ++p) {
                const int row = p * 32 + arr;
                __bf16* d = &As[nb][row * LDW + acc0];
                *(v2bf*)d       = f2bf2(a_s[p].x, a_s[p].y);
                *(v2bf*)(d + 2) = f2bf2(a_s[p].z, a_s[p].w);
            }
#pragma unroll
            for (int p = 0; p < 4; ++p) {
                const int kk = p * 8 + wrr;
                Bs[nb][(wcc + 0) * LDW + kk] = f2bf(w_s[p].x);
                Bs[nb][(wcc + 1) * LDW + kk] = f2bf(w_s[p].y);
                Bs[nb][(wcc + 2) * LDW + kk] = f2bf(w_s[p].z);
                Bs[nb][(wcc + 3) * LDW + kk] = f2bf(w_s[p].w);
            }
        }
        cur ^= 1;
        __syncthreads();
    }

    // Epilogue: C/D layout -> row = base + half*8 + i, col = base + (lane&15)
#pragma unroll
    for (int nt = 0; nt < 4; ++nt) {
        const int col = n0 + waveN * 64 + nt * 16 + r;
        const float bv = bias[col];
#pragma unroll
        for (int mt = 0; mt < 2; ++mt) {
#pragma unroll
            for (int i = 0; i < 8; ++i) {
                const int row = m0 + waveM * 32 + mt * 16 + half * 8 + i;
                float x = acc[mt][nt][i] + bv;
                if (ACT == 1) x = 0.5f * x * (1.0f + erff(x * 0.70710678118654752f));
                C[(size_t)row * N + col] = x;
            }
        }
    }
}

// ---------------------------------------------------------------------------
// Flash attention: one block = (b, h, 64 q-rows). Streams 64-wide KV tiles.
// scores = Q@K^T * scale + bias_matrix*coef + (1-mask)*NEG ; online softmax
// (4 threads per row) ; O += P@V with running rescale. No S x S buffer.
// ---------------------------------------------------------------------------
__global__ __launch_bounds__(256)
void flash_attn_kernel(const float* __restrict__ Qg, const float* __restrict__ Kg,
                       const float* __restrict__ Vg, const float* __restrict__ biasM,
                       const float* __restrict__ bias_coef,
                       const float* __restrict__ mask, float* __restrict__ Og) {
    constexpr int BQ = 64, BKV = 64, LDW = 72, LDS_S = 68;
    __shared__ __align__(16) __bf16 Qs[BQ * LDW];     // [q][dh]
    __shared__ __align__(16) __bf16 Ks[BKV * LDW];    // [kv][dh] (= B^T of K^T)
    __shared__ __align__(16) __bf16 Vts[DH_ * LDW];   // [dh][kv] (= B^T of V)
    __shared__ __align__(16) __bf16 Ps[BQ * LDW];     // probs bf16 [q][kv]
    __shared__ float Ss[BQ * LDS_S];                  // raw scores fp32
    __shared__ float red4[256];
    __shared__ float mrow[BQ], lrow[BQ], frow[BQ];

    const int tid  = threadIdx.x;
    const int wave = tid >> 5, lane = tid & 31;
    const int half = lane >> 4, r = lane & 15;
    const int waveM = wave & 3;   // 4 waves * 16 q-rows
    const int waveN = wave >> 2;  // 2 waves * 32 cols
    const int q0 = blockIdx.x * BQ;
    const int b  = blockIdx.y / NH_;
    const int h  = blockIdx.y % NH_;
    const float bc = bias_coef[0];
    const float scale = 0.125f;   // 1/sqrt(64)

    const int rr = tid >> 4, cc = (tid & 15) << 2;
    // Load Q tile once
#pragma unroll
    for (int p = 0; p < 4; ++p) {
        const int row = p * 16 + rr;
        const float4 f = *(const float4*)&Qg[((size_t)b * S_ + q0 + row) * H_ + h * DH_ + cc];
        __bf16* d = &Qs[row * LDW + cc];
        *(v2bf*)d       = f2bf2(f.x, f.y);
        *(v2bf*)(d + 2) = f2bf2(f.z, f.w);
    }
    if (tid < BQ) { mrow[tid] = -1e30f; lrow[tid] = 0.0f; }

    v8f oacc[2];
#pragma unroll
    for (int nt = 0; nt < 2; ++nt)
#pragma unroll
        for (int i = 0; i < 8; ++i) oacc[nt][i] = 0.0f;

    for (int t = 0; t < S_ / BKV; ++t) {
        __syncthreads();  // previous iteration done with Ks/Vts/Ps
#pragma unroll
        for (int p = 0; p < 4; ++p) {
            const int row = p * 16 + rr;  // kv index in tile
            const size_t base = ((size_t)b * S_ + t * BKV + row) * H_ + h * DH_ + cc;
            const float4 fk = *(const float4*)&Kg[base];
            __bf16* d = &Ks[row * LDW + cc];
            *(v2bf*)d       = f2bf2(fk.x, fk.y);
            *(v2bf*)(d + 2) = f2bf2(fk.z, fk.w);
            const float4 fv = *(const float4*)&Vg[base];
            Vts[(cc + 0) * LDW + row] = f2bf(fv.x);
            Vts[(cc + 1) * LDW + row] = f2bf(fv.y);
            Vts[(cc + 2) * LDW + row] = f2bf(fv.z);
            Vts[(cc + 3) * LDW + row] = f2bf(fv.w);
            if (t + 1 < S_ / BKV) {  // gfx1250 global_prefetch_b8, next KV tile
                __builtin_prefetch(&Kg[base + (size_t)BKV * H_], 0, 1);
                __builtin_prefetch(&Vg[base + (size_t)BKV * H_], 0, 1);
            }
        }
        __syncthreads();

        // S = Q @ K^T  (K-dim = DH = 64 -> 2 WMMA steps)
        v8f sacc[2];
#pragma unroll
        for (int nt = 0; nt < 2; ++nt)
#pragma unroll
            for (int i = 0; i < 8; ++i) sacc[nt][i] = 0.0f;
#pragma unroll
        for (int ks = 0; ks < 2; ++ks) {
            const v16bf af = load_a_frag(Qs, LDW, waveM * 16, ks * 32, half, r);
#pragma unroll
            for (int nt = 0; nt < 2; ++nt) {
                const v16bf bf = load_b_frag(Ks, LDW, waveN * 32 + nt * 16, ks * 32, half, r);
                sacc[nt] = __builtin_amdgcn_wmma_f32_16x16x32_bf16(
                    false, af, false, bf, (short)0, sacc[nt], false, false);
            }
        }
        // scale + pairwise bias + additive mask -> Ss
#pragma unroll
        for (int nt = 0; nt < 2; ++nt) {
            const int col = waveN * 32 + nt * 16 + r;
            const int kg  = t * BKV + col;
            const float mb = (1.0f - mask[(size_t)b * S_ + kg]) * (-10000.0f);
#pragma unroll
            for (int i = 0; i < 8; ++i) {
                const int row = waveM * 16 + half * 8 + i;
                Ss[row * LDS_S + col] =
                    sacc[nt][i] * scale + biasM[(size_t)(q0 + row) * S_ + kg] * bc + mb;
            }
        }
        __syncthreads();

        // Online softmax, 4 threads per q-row (16 cols each)
        {
            const int srow = tid >> 2, ssub = tid & 3;
            float pm = -1e30f;
#pragma unroll
            for (int j = 0; j < 16; ++j)
                pm = fmaxf(pm, Ss[srow * LDS_S + ssub * 16 + j]);
            red4[tid] = pm;
            __syncthreads();
            const float m_old = mrow[srow];
            float m_new = fmaxf(fmaxf(red4[srow * 4 + 0], red4[srow * 4 + 1]),
                                fmaxf(red4[srow * 4 + 2], red4[srow * 4 + 3]));
            m_new = fmaxf(m_new, m_old);
            float psum = 0.0f;
#pragma unroll
            for (int j = 0; j < 16; ++j) {
                const int c = ssub * 16 + j;
                const float p = expf(Ss[srow * LDS_S + c] - m_new);
                Ps[srow * LDW + c] = f2bf(p);
                psum += p;
            }
            __syncthreads();  // red4 max-phase reads done
            red4[tid] = psum;
            __syncthreads();
            if (ssub == 0) {
                const float f = expf(m_old - m_new);
                lrow[srow] = lrow[srow] * f + red4[srow * 4 + 0] + red4[srow * 4 + 1] +
                             red4[srow * 4 + 2] + red4[srow * 4 + 3];
                mrow[srow] = m_new;
                frow[srow] = f;
            }
        }
        __syncthreads();

        // Rescale running O, then O += P @ V
        float fr[8];
#pragma unroll
        for (int i = 0; i < 8; ++i) fr[i] = frow[waveM * 16 + half * 8 + i];
#pragma unroll
        for (int nt = 0; nt < 2; ++nt)
#pragma unroll
            for (int i = 0; i < 8; ++i) oacc[nt][i] *= fr[i];
#pragma unroll
        for (int ks = 0; ks < 2; ++ks) {
            const v16bf af = load_a_frag(Ps, LDW, waveM * 16, ks * 32, half, r);
#pragma unroll
            for (int nt = 0; nt < 2; ++nt) {
                const v16bf bf = load_b_frag(Vts, LDW, waveN * 32 + nt * 16, ks * 32, half, r);
                oacc[nt] = __builtin_amdgcn_wmma_f32_16x16x32_bf16(
                    false, af, false, bf, (short)0, oacc[nt], false, false);
            }
        }
    }
    __syncthreads();

#pragma unroll
    for (int nt = 0; nt < 2; ++nt) {
        const int col = waveN * 32 + nt * 16 + r;  // dh index
#pragma unroll
        for (int i = 0; i < 8; ++i) {
            const int row = waveM * 16 + half * 8 + i;
            Og[((size_t)b * S_ + q0 + row) * H_ + h * DH_ + col] =
                oacc[nt][i] / lrow[row];
        }
    }
}

// ---------------------------------------------------------------------------
// out = LayerNorm(A + R) * g + beta   (block per token row, H = 768 = 256*3)
// ---------------------------------------------------------------------------
__global__ __launch_bounds__(256)
void add_ln_kernel(const float* __restrict__ A, const float* __restrict__ R,
                   const float* __restrict__ g, const float* __restrict__ beta,
                   float* __restrict__ out) {
    __shared__ float red[256];
    const int row = blockIdx.x;
    const int tid = threadIdx.x;

    float y[3];
    float sum = 0.0f;
#pragma unroll
    for (int j = 0; j < 3; ++j) {
        const int col = tid + j * 256;
        const float v = A[(size_t)row * H_ + col] + R[(size_t)row * H_ + col];
        y[j] = v; sum += v;
    }
    red[tid] = sum; __syncthreads();
    for (int s = 128; s > 0; s >>= 1) {
        if (tid < s) red[tid] += red[tid + s];
        __syncthreads();
    }
    const float mu = red[0] * (1.0f / H_);
    float vs = 0.0f;
#pragma unroll
    for (int j = 0; j < 3; ++j) { const float d = y[j] - mu; vs += d * d; }
    __syncthreads();
    red[tid] = vs; __syncthreads();
    for (int s = 128; s > 0; s >>= 1) {
        if (tid < s) red[tid] += red[tid + s];
        __syncthreads();
    }
    const float rstd = rsqrtf(red[0] * (1.0f / H_) + 1e-12f);
#pragma unroll
    for (int j = 0; j < 3; ++j) {
        const int col = tid + j * 256;
        out[(size_t)row * H_ + col] = (y[j] - mu) * rstd * g[col] + beta[col];
    }
}

// ---------------------------------------------------------------------------
extern "C" void kernel_launch(void* const* d_in, const int* in_sizes, int n_in,
                              void* d_out, int out_size, void* d_ws, size_t ws_size,
                              hipStream_t stream) {
    (void)in_sizes; (void)n_in; (void)out_size; (void)ws_size;
    const float* x_in  = (const float*)d_in[0];
    const float* mask  = (const float*)d_in[1];
    const float* biasM = (const float*)d_in[2];
    const float* bcoef = (const float*)d_in[3];
    const float* Wq = (const float*)d_in[4];   const float* bq = (const float*)d_in[5];
    const float* Wk = (const float*)d_in[6];   const float* bk = (const float*)d_in[7];
    const float* Wv = (const float*)d_in[8];   const float* bv = (const float*)d_in[9];
    const float* Wo = (const float*)d_in[10];  const float* bo = (const float*)d_in[11];
    const float* ln1g = (const float*)d_in[12]; const float* ln1b = (const float*)d_in[13];
    const float* Wi = (const float*)d_in[14];  const float* bi = (const float*)d_in[15];
    const float* Wo2 = (const float*)d_in[16]; const float* bo2 = (const float*)d_in[17];
    const float* ln2g = (const float*)d_in[18]; const float* ln2b = (const float*)d_in[19];
    float* out = (float*)d_out;

    // Workspace carve-up (fp32): 7*M*H + M*FF elems ~= 138 MB
    float* ws = (float*)d_ws;
    const size_t MH = (size_t)M_ * H_;
    float* q     = ws;
    float* kbuf  = q + MH;
    float* vbuf  = kbuf + MH;
    float* ctx   = vbuf + MH;
    float* t0    = ctx + MH;
    float* xA    = t0 + MH;
    float* xB    = xA + MH;
    float* inter = xB + MH;  // M * FF

    const dim3 blk(256);
    const dim3 gH(H_ / 128, M_ / 128);    // N = 768
    const dim3 gF(FF_ / 128, M_ / 128);   // N = 3072
    const dim3 gAtt(S_ / 64, B_ * NH_);
    const dim3 gLN(M_);

    const float* xin = x_in;
    for (int i = 0; i < L_; ++i) {
        const size_t oHH = (size_t)i * H_ * H_;
        const size_t oHF = (size_t)i * H_ * FF_;
        gemm_bias_kernel<0><<<gH, blk, 0, stream>>>(xin, Wq + oHH, bq + i * H_, q,    M_, H_, H_);
        gemm_bias_kernel<0><<<gH, blk, 0, stream>>>(xin, Wk + oHH, bk + i * H_, kbuf, M_, H_, H_);
        gemm_bias_kernel<0><<<gH, blk, 0, stream>>>(xin, Wv + oHH, bv + i * H_, vbuf, M_, H_, H_);
        flash_attn_kernel<<<gAtt, blk, 0, stream>>>(q, kbuf, vbuf, biasM, bcoef, mask, ctx);
        gemm_bias_kernel<0><<<gH, blk, 0, stream>>>(ctx, Wo + oHH, bo + i * H_, t0, M_, H_, H_);
        add_ln_kernel<<<gLN, blk, 0, stream>>>(t0, xin, ln1g + i * H_, ln1b + i * H_, xA);
        gemm_bias_kernel<1><<<gF, blk, 0, stream>>>(xA, Wi + oHF, bi + i * FF_, inter, M_, FF_, H_);
        gemm_bias_kernel<0><<<gH, blk, 0, stream>>>(inter, Wo2 + oHF, bo2 + i * H_, t0, M_, H_, FF_);
        float* xo = (i == L_ - 1) ? out : xB;
        add_ln_kernel<<<gLN, blk, 0, stream>>>(t0, xA, ln2g + i * H_, ln2b + i * H_, xo);
        xin = xo;
    }
}
